// WeaveNet_47622597378229
// MI455X (gfx1250) — matrix operand order repaired and checked by reading, not compiled, not run on hardware
//
#include <hip/hip_runtime.h>
#include <hip/hip_bf16.h>

// WeaveNet forward for MI455X (gfx1250, wave32).
// Heavy 1x1 convs -> v_wmma_f32_16x16x32_f16 (f16 in, f32 acc).
// Z kept in f32 (residuals/final head) + f16 mirror Zh (WMMA B operand, halves
// HBM traffic of the dominant GEMM streams). Intermediate z never materialized:
// only its axis-max survives (ordered-uint atomicMax); its conv2 contribution
// is the tiny GEMM t[b,o,idx].

typedef __attribute__((ext_vector_type(16))) _Float16 v16h;
typedef __attribute__((ext_vector_type(8)))  float    v8f;

#define NB 8            // 2*B effective batches
#define HW 65536        // 256*256
#define CH 32           // Z channels after each layer

// order-preserving float<->uint encoding for atomic max (handles negatives)
__device__ __forceinline__ unsigned fenc(float f) {
  unsigned u = __float_as_uint(f);
  return (u & 0x80000000u) ? ~u : (u | 0x80000000u);
}
__device__ __forceinline__ float fdec(unsigned u) {
  return __uint_as_float((u & 0x80000000u) ? (u ^ 0x80000000u) : ~u);
}

// ---- prep: Z0[b,s] = b<4 ? Za[b,s] : Zb[b-4] transposed ----
__global__ void k_prep(const float* __restrict__ Za, const float* __restrict__ Zb,
                       float* __restrict__ Z0) {
  int i = blockIdx.x * blockDim.x + threadIdx.x;     // NB*HW
  if (i >= NB * HW) return;
  int b = i >> 16, s = i & 65535;
  if (b < 4) Z0[i] = Za[(b << 16) + s];
  else {
    int bb = b - 4, h = s >> 8, w = s & 255;
    Z0[i] = Zb[(bb << 16) + (w << 8) + h];
  }
}

// ---- zmax buffer init to enc(-inf) ----
__global__ void k_zmax_init(unsigned* __restrict__ zmaxU) {
  int i = blockIdx.x * blockDim.x + threadIdx.x;     // NB*64*256
  if (i < NB * 64 * 256) zmaxU[i] = 0x007FFFFFu;     // ~bits(-inf)
}

// ---- layer 0 conv1 (2 input ch) + axis max ----
__global__ void k_l0_conv1(const float* __restrict__ Z0, const float* __restrict__ Wm0,
                           const float* __restrict__ bm0, unsigned* __restrict__ zmaxU) {
  int i = blockIdx.x * blockDim.x + threadIdx.x;     // NB*HW
  if (i >= NB * HW) return;
  int b = i >> 16, s = i & 65535;
  float x0 = Z0[i];
  float x1 = Z0[(((b + 4) & 7) << 16) + s];
  int idx = (b < 4) ? (s & 255) : (s >> 8);
  #pragma unroll 4
  for (int o = 0; o < 64; ++o) {
    float z = Wm0[o * 2] * x0 + Wm0[o * 2 + 1] * x1 + bm0[o];
    atomicMax(&zmaxU[(((b << 6) + o) << 8) + idx], fenc(z));
  }
}

// ---- t[b,o,idx] = sum_c Wz[o*stride+c] * zmax[b,c,idx]  (max-half of conv2) ----
__global__ void k_tmat(const unsigned* __restrict__ zmaxU, const float* __restrict__ Wz,
                       int stride, float* __restrict__ t) {
  int i = blockIdx.x * blockDim.x + threadIdx.x;     // NB*32*256
  if (i >= NB * CH * 256) return;
  int idx = i & 255, o = (i >> 8) & 31, b = i >> 13;
  float acc = 0.f;
  #pragma unroll 4
  for (int c = 0; c < 64; ++c)
    acc += Wz[o * stride + c] * fdec(zmaxU[(((b << 6) + c) << 8) + idx]);
  t[i] = acc;
}

// ---- layer 0 conv2 (2 ch + t) ----
__global__ void k_l0_conv2(const float* __restrict__ Z0, const float* __restrict__ W0,
                           const float* __restrict__ t, float* __restrict__ Y) {
  int i = blockIdx.x * blockDim.x + threadIdx.x;     // NB*HW
  if (i >= NB * HW) return;
  int b = i >> 16, s = i & 65535;
  float x0 = Z0[i];
  float x1 = Z0[(((b + 4) & 7) << 16) + s];
  int idx = (b < 4) ? (s & 255) : (s >> 8);
  #pragma unroll 4
  for (int o = 0; o < 32; ++o)
    Y[(((b << 5) + o) << 16) + s] =
        W0[o * 66] * x0 + W0[o * 66 + 1] * x1 + t[(((b << 5) + o) << 8) + idx];
}

// ---- conv1 WMMA (layers 1..5): z = Wm(64x64) * X(64 x spatial) + bm; axis-max ----
// A fragments hoisted; each wave sweeps 8 spatial tiles. K-step 0 touches only
// batch b channels, K-step 1 only batch b2 channels (cross-concat structure).
__global__ __launch_bounds__(128) void k_conv1_wmma(
    const _Float16* __restrict__ Zh, // [8,32,65536] f16 mirror
    const float* __restrict__ Wm,    // [64,64]
    const float* __restrict__ bm,    // [64]
    unsigned* __restrict__ zmaxU) {  // [8,64,256] encoded
  int lane = threadIdx.x & 31;
  int wave = threadIdx.x >> 5;
  int blk  = blockIdx.x;           // NB * 512
  int b    = blk >> 9;
  int sbase = (blk & 511) << 7;    // 128 spatial per block
  int o0   = wave << 4;
  int hi   = (lane >= 16) ? 8 : 0;
  int nn   = lane & 15;
  int b2   = (b + 4) & 7;

  // A fragments (contiguous K: k = 16*(lane>=16) + h), loop-invariant
  v16h a0, a1;
  {
    int m  = o0 + nn;
    int ka = (lane >= 16) ? 16 : 0;
    #pragma unroll
    for (int h = 0; h < 16; ++h) {
      int k = ka + h;
      a0[h] = (_Float16)Wm[m * 64 + k];
      a1[h] = (_Float16)Wm[m * 64 + 32 + k];
    }
  }
  int kb0 = (lane >= 16) ? 16 : 0;                   // B lane-half K split
  const _Float16* pb  = Zh + (((b  << 5) + kb0) << 16);
  const _Float16* pb2 = Zh + (((b2 << 5) + kb0) << 16);

  for (int tt = 0; tt < 8; ++tt) {
    int s = sbase + (tt << 4) + nn;
    if (tt < 7) __builtin_prefetch(pb + s + 16, 0, 0);

    v8f acc;
    #pragma unroll
    for (int r = 0; r < 8; ++r) acc[r] = bm[o0 + hi + r];

    v16h bv0, bv1;
    #pragma unroll
    for (int h = 0; h < 16; ++h) {
      bv0[h] = pb [(h << 16) + s];
      bv1[h] = pb2[(h << 16) + s];
    }
    acc = __builtin_amdgcn_wmma_f32_16x16x32_f16(false, a0, false, bv0, (short)0, acc,
                                                 false, false);
    acc = __builtin_amdgcn_wmma_f32_16x16x32_f16(false, a1, false, bv1, (short)0, acc,
                                                 false, false);

    int idx = (b < 4) ? (s & 255) : (s >> 8);
    #pragma unroll
    for (int r = 0; r < 8; ++r) {
      int o = o0 + hi + r;                 // D: M = r + 8*(lane>=16), N = lane&15
      atomicMax(&zmaxU[(((b << 6) + o) << 8) + idx], fenc(acc[r]));
    }
  }
}

// ---- conv2 WMMA (layers 1..5): Y = W(32x64) * X(64 x spatial) + t[b,o,idx] ----
__global__ __launch_bounds__(128) void k_conv2_wmma(
    const _Float16* __restrict__ Zh, // [8,32,65536] f16 mirror
    const float* __restrict__ W,     // [32,128], cols 0..63 used here
    const float* __restrict__ t,     // [8,32,256]
    float* __restrict__ Y) {         // [8,32,65536]
  int lane = threadIdx.x & 31;
  int wave = threadIdx.x >> 5;
  int blk  = blockIdx.x;           // NB * 256
  int b    = blk >> 8;
  int sbase = (blk & 255) << 8;    // 256 spatial per block
  int o0   = (wave & 1) << 4;
  int sub  = wave >> 1;
  int hi   = (lane >= 16) ? 8 : 0;
  int nn   = lane & 15;
  int b2   = (b + 4) & 7;

  v16h a0, a1;
  {
    int m  = o0 + nn;
    int ka = (lane >= 16) ? 16 : 0;
    #pragma unroll
    for (int h = 0; h < 16; ++h) {
      int k = ka + h;
      a0[h] = (_Float16)W[m * 128 + k];
      a1[h] = (_Float16)W[m * 128 + 32 + k];
    }
  }
  int kb0 = (lane >= 16) ? 16 : 0;
  const _Float16* pb  = Zh + (((b  << 5) + kb0) << 16);
  const _Float16* pb2 = Zh + (((b2 << 5) + kb0) << 16);

  for (int tt = 0; tt < 8; ++tt) {
    int s = sbase + (tt << 5) + (sub << 4) + nn;
    if (tt < 7) __builtin_prefetch(pb + s + 32, 0, 0);
    int idx = (b < 4) ? (s & 255) : (s >> 8);

    v8f acc;
    #pragma unroll
    for (int r = 0; r < 8; ++r)
      acc[r] = t[(((b << 5) + o0 + hi + r) << 8) + idx];

    v16h bv0, bv1;
    #pragma unroll
    for (int h = 0; h < 16; ++h) {
      bv0[h] = pb [(h << 16) + s];
      bv1[h] = pb2[(h << 16) + s];
    }
    acc = __builtin_amdgcn_wmma_f32_16x16x32_f16(false, a0, false, bv0, (short)0, acc,
                                                 false, false);
    acc = __builtin_amdgcn_wmma_f32_16x16x32_f16(false, a1, false, bv1, (short)0, acc,
                                                 false, false);

    #pragma unroll
    for (int r = 0; r < 8; ++r)
      Y[(((b << 5) + o0 + hi + r) << 16) + s] = acc[r];
  }
}

// ---- BN stats per channel (deterministic two-level reduction) ----
__global__ __launch_bounds__(256) void k_bnstats(const float* __restrict__ Y,
                                                 float* __restrict__ meanArr,
                                                 float* __restrict__ rstdArr) {
  __shared__ float ssum[256], ssq[256];
  int ch = blockIdx.x, tid = threadIdx.x;
  float s = 0.f, q = 0.f;
  for (int b = 0; b < NB; ++b) {
    const float* p = Y + (((b << 5) + ch) << 16);
    for (int i = tid; i < HW; i += 256) { float v = p[i]; s += v; q += v * v; }
  }
  ssum[tid] = s; ssq[tid] = q; __syncthreads();
  for (int st = 128; st > 0; st >>= 1) {
    if (tid < st) { ssum[tid] += ssum[tid + st]; ssq[tid] += ssq[tid + st]; }
    __syncthreads();
  }
  if (tid == 0) {
    float n   = (float)(NB * HW);
    float mu  = ssum[0] / n;
    float var = ssq[0] / n - mu * mu;
    meanArr[ch] = mu;
    rstdArr[ch] = rsqrtf(var + 1e-5f);
  }
}

// ---- BN apply + PReLU + residual; writes f32 Z and f16 mirror Zh ----
__global__ void k_bnapply(const float* __restrict__ Y, const float* __restrict__ meanArr,
                          const float* __restrict__ rstdArr, const float* __restrict__ g,
                          const float* __restrict__ be, const float* __restrict__ aP,
                          const float* __restrict__ ZkIn, float* __restrict__ Z,
                          _Float16* __restrict__ Zh, float* __restrict__ ZkOut,
                          int addRes, int writeKeep) {
  int i = blockIdx.x * blockDim.x + threadIdx.x;     // NB*CH*HW = 16777216
  if (i >= NB * CH * HW) return;
  int c = (i >> 16) & 31;
  float v = (Y[i] - meanArr[c]) * rstdArr[c] * g[c] + be[c];
  float a = aP[0];
  v = (v >= 0.f) ? v : a * v;
  if (addRes) v += ZkIn[i];
  Z[i] = v;
  Zh[i] = (_Float16)v;
  if (writeKeep) ZkOut[i] = v;
}

// ---- final 1x1 conv over cross-concat(Z): O=1 ----
__global__ void k_final_conv(const float* __restrict__ Z, const float* __restrict__ Wf,
                             float* __restrict__ Zc) {
  int i = blockIdx.x * blockDim.x + threadIdx.x;     // NB*HW
  if (i >= NB * HW) return;
  int b = i >> 16, s = i & 65535, b2 = (b + 4) & 7;
  float acc = 0.f;
  #pragma unroll 4
  for (int c = 0; c < 32; ++c)
    acc += Wf[c]      * Z[(((b  << 5) + c) << 16) + s]
         + Wf[32 + c] * Z[(((b2 << 5) + c) << 16) + s];
  Zc[i] = acc;
}

// ---- final BN stats (single channel, one block) ----
__global__ __launch_bounds__(256) void k_final_stats(const float* __restrict__ Zc,
                                                     float* __restrict__ stF) {
  __shared__ float ssum[256], ssq[256];
  int tid = threadIdx.x;
  float s = 0.f, q = 0.f;
  for (int i = tid; i < NB * HW; i += 256) { float v = Zc[i]; s += v; q += v * v; }
  ssum[tid] = s; ssq[tid] = q; __syncthreads();
  for (int st = 128; st > 0; st >>= 1) {
    if (tid < st) { ssum[tid] += ssum[tid + st]; ssq[tid] += ssq[tid + st]; }
    __syncthreads();
  }
  if (tid == 0) {
    float n   = (float)(NB * HW);
    float mu  = ssum[0] / n;
    float var = ssq[0] / n - mu * mu;
    stF[0] = mu;
    stF[1] = rsqrtf(var + 1e-5f);
  }
}

// ---- m0/m1 (to out) + mm buffer ----
__global__ void k_mm(const float* __restrict__ Zc, const float* __restrict__ stF,
                     const float* __restrict__ gf, const float* __restrict__ bf,
                     float* __restrict__ out, float* __restrict__ mmbuf) {
  int i = blockIdx.x * blockDim.x + threadIdx.x;     // 4*HW
  if (i >= 4 * HW) return;
  int b = i >> 16, s = i & 65535;
  float mu = stF[0], rs = stF[1], G = gf[0], Bv = bf[0];
  float m0 = (Zc[(b << 16) + s]       - mu) * rs * G + Bv;
  float m1 = (Zc[((b + 4) << 16) + s] - mu) * rs * G + Bv;
  out[262144 + i] = m0;
  out[524288 + i] = m1;
  mmbuf[i] = 0.5f * (m0 + m1);
}

// ---- softmax over w (rows) ----
__global__ __launch_bounds__(256) void k_softmax_row(const float* __restrict__ mm,
                                                     float* __restrict__ rowsm) {
  __shared__ float sh[256];
  int row = blockIdx.x;                // b*256 + h
  int w = threadIdx.x;
  float v = mm[(row << 8) + w];
  sh[w] = v; __syncthreads();
  for (int st = 128; st > 0; st >>= 1) {
    if (w < st) sh[w] = fmaxf(sh[w], sh[w + st]);
    __syncthreads();
  }
  float mx = sh[0]; __syncthreads();
  float e = expf(v - mx);
  sh[w] = e; __syncthreads();
  for (int st = 128; st > 0; st >>= 1) {
    if (w < st) sh[w] += sh[w + st];
    __syncthreads();
  }
  rowsm[(row << 8) + w] = e / sh[0];
}

// ---- softmax over h (cols) + agg = min(row_sm, col_sm) ----
__global__ __launch_bounds__(256) void k_softmax_col(const float* __restrict__ mm,
                                                     const float* __restrict__ rowsm,
                                                     float* __restrict__ out) {
  __shared__ float sh[256];
  int col = blockIdx.x;                // b*256 + w
  int b = col >> 8, w = col & 255;
  int h = threadIdx.x;
  int ix = (b << 16) + (h << 8) + w;
  float v = mm[ix];
  sh[h] = v; __syncthreads();
  for (int st = 128; st > 0; st >>= 1) {
    if (h < st) sh[h] = fmaxf(sh[h], sh[h + st]);
    __syncthreads();
  }
  float mx = sh[0]; __syncthreads();
  float e = expf(v - mx);
  sh[h] = e; __syncthreads();
  for (int st = 128; st > 0; st >>= 1) {
    if (h < st) sh[h] += sh[h + st];
    __syncthreads();
  }
  out[ix] = fminf(rowsm[ix], e / sh[0]);
}

extern "C" void kernel_launch(void* const* d_in, const int* in_sizes, int n_in,
                              void* d_out, int out_size, void* d_ws, size_t ws_size,
                              hipStream_t stream) {
  const float* Za  = (const float*)d_in[0];
  const float* Zb  = (const float*)d_in[1];
  const float* Wm0 = (const float*)d_in[2];
  const float* bm0 = (const float*)d_in[3];
  const float* W0  = (const float*)d_in[4];
  const float* g0  = (const float*)d_in[5];
  const float* be0 = (const float*)d_in[6];
  const float* Wm  = (const float*)d_in[7];   // (5,64,64)
  const float* bm  = (const float*)d_in[8];   // (5,64)
  const float* Wl  = (const float*)d_in[9];   // (5,32,128)
  const float* g   = (const float*)d_in[10];  // (5,32)
  const float* be  = (const float*)d_in[11];  // (5,32)
  const float* aP  = (const float*)d_in[12];  // scalar
  const float* Wf  = (const float*)d_in[13];  // (1,64)
  const float* gf  = (const float*)d_in[14];
  const float* bf  = (const float*)d_in[15];
  float* out = (float*)d_out;

  // workspace layout (float units unless noted)
  float*    Zbuf    = (float*)d_ws;                 // 16777216
  float*    Zk      = Zbuf + 16777216;              // 16777216
  float*    Ybuf    = Zk + 16777216;                // 16777216
  float*    Z0      = Ybuf + 16777216;              // 524288 (reused as Zc)
  unsigned* zmaxU   = (unsigned*)(Z0 + 524288);     // 131072
  float*    tbuf    = (float*)(zmaxU + 131072);     // 65536
  float*    meanArr = tbuf + 65536;                 // 32
  float*    rstdArr = meanArr + 32;                 // 32
  float*    stF     = rstdArr + 32;                 // 2
  float*    mmbuf   = stF + 2;                      // 262144
  float*    rowsm   = mmbuf + 262144;               // 262144
  _Float16* Zh      = (_Float16*)(rowsm + 262144);  // 16777216 halves

  k_prep<<<2048, 256, 0, stream>>>(Za, Zb, Z0);

  // ---- layer 0 ----
  k_zmax_init<<<512, 256, 0, stream>>>(zmaxU);
  k_l0_conv1<<<2048, 256, 0, stream>>>(Z0, Wm0, bm0, zmaxU);
  k_tmat<<<256, 256, 0, stream>>>(zmaxU, W0 + 2, 66, tbuf);
  k_l0_conv2<<<2048, 256, 0, stream>>>(Z0, W0, tbuf, Ybuf);
  k_bnstats<<<32, 256, 0, stream>>>(Ybuf, meanArr, rstdArr);
  k_bnapply<<<65536, 256, 0, stream>>>(Ybuf, meanArr, rstdArr, g0, be0, aP,
                                       Zk, Zbuf, Zh, Zk, /*addRes=*/0, /*writeKeep=*/1);

  // ---- layers 1..5 ----
  for (int i = 1; i < 6; ++i) {
    const float* Wm_i = Wm + (i - 1) * 64 * 64;
    const float* bm_i = bm + (i - 1) * 64;
    const float* W_i  = Wl + (i - 1) * 32 * 128;
    const float* g_i  = g  + (i - 1) * 32;
    const float* be_i = be + (i - 1) * 32;
    int addRes    = (i == 2 || i == 4);
    int writeKeep = (i == 2 || i == 4);

    k_zmax_init<<<512, 256, 0, stream>>>(zmaxU);
    k_conv1_wmma<<<NB * 512, 128, 0, stream>>>(Zh, Wm_i, bm_i, zmaxU);
    k_tmat<<<256, 256, 0, stream>>>(zmaxU, W_i + 64, 128, tbuf);
    k_conv2_wmma<<<NB * 256, 128, 0, stream>>>(Zh, W_i, tbuf, Ybuf);
    k_bnstats<<<32, 256, 0, stream>>>(Ybuf, meanArr, rstdArr);
    k_bnapply<<<65536, 256, 0, stream>>>(Ybuf, meanArr, rstdArr, g_i, be_i, aP,
                                         Zk, Zbuf, Zh, Zk, addRes, writeKeep);
  }

  // ---- final head ----
  k_final_conv<<<2048, 256, 0, stream>>>(Zbuf, Wf, Z0);   // Z0 reused as Zc
  k_final_stats<<<1, 256, 0, stream>>>(Z0, stF);
  k_mm<<<1024, 256, 0, stream>>>(Z0, stF, gf, bf, out, mmbuf);
  k_softmax_row<<<1024, 256, 0, stream>>>(mmbuf, rowsm);
  k_softmax_col<<<1024, 256, 0, stream>>>(mmbuf, rowsm, out);
}
